// MultiHeadAttentionBlock_9809705304091
// MI455X (gfx1250) — matrix operand order repaired and checked by reading, not compiled, not run on hardware
//
#include <hip/hip_runtime.h>
#include <hip/hip_bf16.h>
#include <math.h>

#define B_  2
#define S_  2048
#define D_  1024
#define H_  16
#define DH_ 64

typedef __attribute__((ext_vector_type(16))) _Float16 v16h;
typedef __attribute__((ext_vector_type(8)))  _Float16 v8h;
typedef __attribute__((ext_vector_type(8)))  float    v8f;

// ---- WMMA helper: D = A(16x32 f16) * B(32x16 f16) + C(16x16 f32) ----
__device__ __forceinline__ v8f wmma_f16(v16h a, v16h b, v8f c) {
  return __builtin_amdgcn_wmma_f32_16x16x32_f16(false, a, false, b, (short)0, c,
                                                false, false);
}

// A-fragment (16x32, 16-bit): lane L holds row m=L%16; half h=L/16 holds
// K chunks [h*8, h*8+8) in v[0..7] and [16+h*8, ...+8) in v[8..15].
__device__ __forceinline__ v16h load_fragA(const _Float16* base, int ld, int m, int h) {
  const _Float16* r = base + (size_t)m * ld;
  v8h lo = *(const v8h*)(r + h * 8);
  v8h hi = *(const v8h*)(r + 16 + h * 8);
  v16h o;
#pragma unroll
  for (int i = 0; i < 8; i++) { o[i] = lo[i]; o[i + 8] = hi[i]; }
  return o;
}

// B-fragment (32x16, 16-bit): lane L holds column n=L%16; half h=L/16 holds
// K chunks [h*8, h*8+8) in v[0..7] and [16+h*8, ...+8) in v[8..15]. "base" = row 0 of the pre-transposed matrix.
__device__ __forceinline__ v16h load_fragB(const _Float16* base, int ld, int n, int h) {
  const _Float16* r = base + (size_t)n * ld;
  v8h lo = *(const v8h*)(r + h * 8);
  v8h hi = *(const v8h*)(r + 16 + h * 8);
  v16h o;
#pragma unroll
  for (int i = 0; i < 8; i++) { o[i] = lo[i]; o[i + 8] = hi[i]; }
  return o;
}

// ---------------- Tensor Data Mover staging (gfx1250) ----------------
#if __has_builtin(__builtin_amdgcn_tensor_load_to_lds)
#define HAVE_TDM 1
typedef __attribute__((ext_vector_type(4))) unsigned int v4u;
typedef __attribute__((ext_vector_type(4))) int v4i;
typedef __attribute__((ext_vector_type(8))) int v8i;

// 2-D tile (tile_rows x tile_cols elements of 2 bytes) from row-major global
// (row pitch ld_elems) into contiguous LDS at byte offset lds_off.
// D# packing per CDNA5 ISA 8.3/8.4 (count=1, type=2, data_size=1 -> 2B).
__device__ __forceinline__ void tdm_load_tile_2d(const void* gsrc, unsigned lds_off,
                                                 unsigned tile_cols, unsigned tile_rows,
                                                 unsigned ld_elems) {
  unsigned long long ga = (unsigned long long)(size_t)gsrc;
  v4u g0;
  g0[0] = 1u;                                    // count=1 (valid), user mode
  g0[1] = lds_off;                               // lds_addr (bytes)
  g0[2] = (unsigned)ga;                          // global_addr[31:0]  -> bits 95:64
  g0[3] = (unsigned)((ga >> 32) & 0x01ffffffu)   // global_addr[56:32] -> bits 120:96
          | (2u << 30);                          // type = 2 ("image")
  v8i g1;
  g1[0] = (int)(1u << 16);                       // data_size = 1 (2-byte elements)
  g1[1] = (int)((ld_elems & 0xffffu) << 16);     // tensor_dim0[15:0] @ bits 63:48
  g1[2] = (int)(((ld_elems >> 16) & 0xffffu)     // tensor_dim0[31:16]
          | ((tile_rows & 0xffffu) << 16));      // tensor_dim1[15:0]
  g1[3] = (int)((tile_cols & 0xffffu) << 16);    // tensor_dim1[31:16]=0 | tile_dim0
  g1[4] = (int)(tile_rows & 0xffffu);            // tile_dim1 | tile_dim2=0
  g1[5] = (int)ld_elems;                         // tensor_dim0_stride[31:0]
  g1[6] = 0;                                     // stride[47:32]=0 | dim1_stride lo
  g1[7] = 0;
  v4i gz = (v4i)0;                               // groups 2/3 unused (2-D tile)
#if defined(__clang_major__) && __clang_major__ >= 23
  v8i gz8 = (v8i)0;
  __builtin_amdgcn_tensor_load_to_lds(g0, g1, gz, gz, gz8, 0);
#else
  __builtin_amdgcn_tensor_load_to_lds(g0, g1, gz, gz, 0);
#endif
}
#else
#define HAVE_TDM 0
#endif

__global__ void cvt_f32_to_f16(const float* __restrict__ src,
                               _Float16* __restrict__ dst, int n) {
  int i = blockIdx.x * blockDim.x + threadIdx.x;
  int stride = gridDim.x * blockDim.x;
  for (; i < n; i += stride) dst[i] = (_Float16)src[i];
}

// Y = X @ W^T + bias. X: [4096,1024] f16, W: [1024,1024] f16 row-major
// (row n of W == column n of the WMMA B matrix: contiguous).
// Block: 8 waves, 256(M) x 64(N) tile; wave = 32(M) x 64(N), 8 WMMAs / K-step.
// The shared 64x32 weight tile is DMA'd into LDS by the TDM (double-buffered)
// so all 8 waves read it via ds_load instead of 8x redundant global traffic.
// mode 0: f16 -> [B,H,S,DH] * scale | mode 1: f16 -> [B,H,DH,S] | mode 2: f32.
__global__ __launch_bounds__(256)
void proj_gemm16(const _Float16* __restrict__ X, const _Float16* __restrict__ W,
                 const float* __restrict__ bias, void* __restrict__ out,
                 int mode, float scale) {
  __shared__ _Float16 Wlds[2][64 * 32];          // 8 KB double-buffered B tile

  const int lane = threadIdx.x & 31;
  const int wv   = threadIdx.x >> 5;
  const int half = lane >> 4;
  const int ln16 = lane & 15;
  const int rbase = blockIdx.x * 256 + wv * 32;  // 32 rows per wave
  const int nbase = blockIdx.y * 64;

  v8f acc[2][4];
#pragma unroll
  for (int j = 0; j < 4; j++) {
    float bj = bias[nbase + 16 * j + ln16];
#pragma unroll
    for (int r = 0; r < 8; r++) { acc[0][j][r] = bj; acc[1][j][r] = bj; }
  }

  const _Float16* Wtile = W + (size_t)nbase * D_;      // 64 rows of W
  const _Float16* X0 = X + (size_t)rbase * D_;
  const _Float16* X1 = X + (size_t)(rbase + 16) * D_;

  const int NIT = D_ / 32;

#if HAVE_TDM
  if (wv == 0)
    tdm_load_tile_2d(Wtile, (unsigned)(size_t)(void*)&Wlds[0][0], 32, 64, D_);
#endif
  // register-prefetched A fragments for kb = 0
  v16h a0 = load_fragA(X0, D_, ln16, half);
  v16h a1 = load_fragA(X1, D_, ln16, half);

  for (int it = 0; it < NIT; ++it) {
    const int kb = it * 32;
    const bool more = (it + 1 < NIT);
    const int cur = it & 1;

#if HAVE_TDM
    if (wv == 0) {
      if (more) {  // overlap next tile's DMA with this tile's compute
        tdm_load_tile_2d(Wtile + kb + 32,
                         (unsigned)(size_t)(void*)&Wlds[cur ^ 1][0], 32, 64, D_);
        __builtin_amdgcn_s_wait_tensorcnt(1);   // current tile resident, next in flight
      } else {
        __builtin_amdgcn_s_wait_tensorcnt(0);   // last tile resident
      }
    }
    __syncthreads();
#else
    for (int idx = threadIdx.x; idx < 64 * 32; idx += 256)
      Wlds[0][idx] = Wtile[(size_t)(idx >> 5) * D_ + kb + (idx & 31)];
    __syncthreads();
#endif
    const _Float16* Bt = &Wlds[HAVE_TDM ? cur : 0][0];

    // prefetch next A fragments before consuming the current ones
    v16h na0, na1;
    if (more) {
      na0 = load_fragA(X0 + kb + 32, D_, ln16, half);
      na1 = load_fragA(X1 + kb + 32, D_, ln16, half);
    }
#pragma unroll
    for (int j = 0; j < 4; j++) {
      v16h b = load_fragB(Bt + (size_t)(16 * j) * 32, 32, ln16, half);
      acc[0][j] = wmma_f16(a0, b, acc[0][j]);
      acc[1][j] = wmma_f16(a1, b, acc[1][j]);
    }
    __syncthreads();   // readers done before buffer is re-DMA'd
    if (more) { a0 = na0; a1 = na1; }
  }

#pragma unroll
  for (int s = 0; s < 2; ++s) {
#pragma unroll
    for (int j = 0; j < 4; j++) {
#pragma unroll
      for (int r = 0; r < 8; r++) {
        int mrow = rbase + 16 * s + r + 8 * half;
        int col  = nbase + 16 * j + ln16;
        float v = acc[s][j][r];
        if (mode == 2) {
          ((float*)out)[(size_t)mrow * D_ + col] = v;
        } else {
          v *= scale;
          int b = mrow / S_, ss = mrow % S_;
          int hh = col / DH_, d = col % DH_;
          _Float16 hv = (_Float16)v;
          if (mode == 0)
            ((_Float16*)out)[(((size_t)(b * H_ + hh)) * S_ + ss) * DH_ + d] = hv;
          else
            ((_Float16*)out)[(((size_t)(b * H_ + hh)) * DH_ + d) * S_ + ss] = hv;
        }
      }
    }
  }
}

// Flash attention: Q[B,H,S,DH] f16 (pre-scaled by 1/sqrt(DH)), K[B,H,S,DH] f16,
// VT[B,H,DH,S] f16, mask[B,1,S,S] i32 -> f16 row-major [B*S, D].
// K B-fragments are double-buffered in registers across the kv loop.
__global__ __launch_bounds__(256)
void flash_attn(const _Float16* __restrict__ Q, const _Float16* __restrict__ K,
                const _Float16* __restrict__ VT, const int* __restrict__ mask,
                _Float16* __restrict__ O16) {
  __shared__ _Float16 Plds[8][16][32];   // per-wave private P staging (C->A relayout)

  const int lane = threadIdx.x & 31;
  const int wv   = threadIdx.x >> 5;
  const int half = lane >> 4;
  const int ln16 = lane & 15;
  const int bh = blockIdx.y;
  const int b  = bh / H_;
  const int hh = bh % H_;
  const int qbase = blockIdx.x * 128 + wv * 16;

  const _Float16* Qrow   = Q  + ((size_t)bh * S_ + qbase) * DH_;
  const _Float16* Kbase  = K  + (size_t)bh * S_ * DH_;
  const _Float16* VTbase = VT + (size_t)bh * DH_ * S_;
  const int*      mrow   = mask + (size_t)b * S_ * S_;

  v16h qa0 = load_fragA(Qrow + 0,  DH_, ln16, half);
  v16h qa1 = load_fragA(Qrow + 32, DH_, ln16, half);

  v8f acc[4];
#pragma unroll
  for (int j = 0; j < 4; j++)
#pragma unroll
    for (int r = 0; r < 8; r++) acc[j][r] = 0.f;

  float m_run[8], l_run[8];
#pragma unroll
  for (int r = 0; r < 8; r++) { m_run[r] = -3.0e38f; l_run[r] = 0.f; }

  // prologue: K fragments for kb = 0
  v16h kf[4];
  kf[0] = load_fragB(Kbase + 0,             DH_, ln16, half);
  kf[1] = load_fragB(Kbase + 32,            DH_, ln16, half);
  kf[2] = load_fragB(Kbase + 16 * DH_ + 0,  DH_, ln16, half);
  kf[3] = load_fragB(Kbase + 16 * DH_ + 32, DH_, ln16, half);

  for (int kb = 0; kb < S_; kb += 32) {
    const bool more = (kb + 32 < S_);
    // prefetch next iteration's K fragments
    v16h nk[4];
    if (more) {
      const _Float16* Kn = Kbase + (size_t)(kb + 32) * DH_;
      nk[0] = load_fragB(Kn + 0,             DH_, ln16, half);
      nk[1] = load_fragB(Kn + 32,            DH_, ln16, half);
      nk[2] = load_fragB(Kn + 16 * DH_ + 0,  DH_, ln16, half);
      nk[3] = load_fragB(Kn + 16 * DH_ + 32, DH_, ln16, half);
    }
    // ---- scores: two 16x16 tiles ----
    v8f s0, s1;
#pragma unroll
    for (int r = 0; r < 8; r++) { s0[r] = 0.f; s1[r] = 0.f; }
    s0 = wmma_f16(qa0, kf[0], s0);
    s0 = wmma_f16(qa1, kf[1], s0);
    s1 = wmma_f16(qa0, kf[2], s1);
    s1 = wmma_f16(qa1, kf[3], s1);
    // ---- mask ----
#pragma unroll
    for (int r = 0; r < 8; r++) {
      int q = qbase + r + 8 * half;
      if (mrow[(size_t)q * S_ + kb + ln16]      == 0) s0[r] = -1e9f;
      if (mrow[(size_t)q * S_ + kb + 16 + ln16] == 0) s1[r] = -1e9f;
    }
    // ---- online softmax (row stats stay inside each 16-lane half) ----
    float alph[8];
#pragma unroll
    for (int r = 0; r < 8; r++) {
      float v = fmaxf(s0[r], s1[r]);
      v = fmaxf(v, __shfl_xor(v, 1));
      v = fmaxf(v, __shfl_xor(v, 2));
      v = fmaxf(v, __shfl_xor(v, 4));
      v = fmaxf(v, __shfl_xor(v, 8));
      float mnew = fmaxf(m_run[r], v);
      float a  = __expf(m_run[r] - mnew);
      float p0 = __expf(s0[r] - mnew);
      float p1 = __expf(s1[r] - mnew);
      s0[r] = p0; s1[r] = p1;
      float ps = p0 + p1;
      ps += __shfl_xor(ps, 1);
      ps += __shfl_xor(ps, 2);
      ps += __shfl_xor(ps, 4);
      ps += __shfl_xor(ps, 8);
      m_run[r] = mnew;
      l_run[r] = l_run[r] * a + ps;
      alph[r] = a;
    }
#pragma unroll
    for (int j = 0; j < 4; j++)
#pragma unroll
      for (int r = 0; r < 8; r++) acc[j][r] *= alph[r];

    // ---- P: C-layout regs -> LDS -> A-layout frag (per-wave private slice) ----
#pragma unroll
    for (int r = 0; r < 8; r++) {
      int row = r + 8 * half;
      Plds[wv][row][ln16]      = (_Float16)s0[r];
      Plds[wv][row][16 + ln16] = (_Float16)s1[r];
    }
    v16h pa = load_fragA(&Plds[wv][0][0], 32, ln16, half);

    // ---- O += P(16x32) @ V(32x64) ----
#pragma unroll
    for (int j = 0; j < 4; j++) {
      v16h vb = load_fragB(VTbase + (size_t)(16 * j) * S_ + kb, S_, ln16, half);
      acc[j] = wmma_f16(pa, vb, acc[j]);
    }
    if (more) {
#pragma unroll
      for (int t = 0; t < 4; t++) kf[t] = nk[t];
    }
  }

#pragma unroll
  for (int j = 0; j < 4; j++)
#pragma unroll
    for (int r = 0; r < 8; r++) {
      int q = qbase + r + 8 * half;
      int d = hh * DH_ + 16 * j + ln16;
      O16[((size_t)b * S_ + q) * D_ + d] = (_Float16)(acc[j][r] / l_run[r]);
    }
}

extern "C" void kernel_launch(void* const* d_in, const int* in_sizes, int n_in,
                              void* d_out, int out_size, void* d_ws, size_t ws_size,
                              hipStream_t stream) {
  const float* query = (const float*)d_in[0];
  const float* key   = (const float*)d_in[1];
  const float* value = (const float*)d_in[2];
  const int*   mask  = (const int*)d_in[3];
  const float* Wq = (const float*)d_in[4];
  const float* bq = (const float*)d_in[5];
  const float* Wk = (const float*)d_in[6];
  const float* bk = (const float*)d_in[7];
  const float* Wv = (const float*)d_in[8];
  const float* bv = (const float*)d_in[9];
  const float* Wo = (const float*)d_in[10];
  const float* bo = (const float*)d_in[11];

  char* ws = (char*)d_ws;
  const size_t nX = (size_t)B_ * S_ * D_;
  const size_t nW = (size_t)D_ * D_;
  _Float16* x16  = (_Float16*)ws; ws += nX * 2;
  _Float16* w16  = (_Float16*)ws; ws += nW * 2;
  _Float16* q16  = (_Float16*)ws; ws += nX * 2;   // [B,H,S,DH], pre-scaled
  _Float16* k16  = (_Float16*)ws; ws += nX * 2;   // [B,H,S,DH]
  _Float16* vT16 = (_Float16*)ws; ws += nX * 2;   // [B,H,DH,S]
  _Float16* a16  = (_Float16*)ws; ws += nX * 2;   // attention out, [B*S, D]

  dim3 cvtg(1024), cvtb(256);
  dim3 pgrid(B_ * S_ / 256, D_ / 64), pblk(256);
  dim3 agrid(S_ / 128, B_ * H_), ablk(256);

  cvt_f32_to_f16<<<cvtg, cvtb, 0, stream>>>(query, x16, (int)nX);
  cvt_f32_to_f16<<<cvtg, cvtb, 0, stream>>>(Wq, w16, (int)nW);
  proj_gemm16<<<pgrid, pblk, 0, stream>>>(x16, w16, bq, (void*)q16, 0, 0.125f);

  cvt_f32_to_f16<<<cvtg, cvtb, 0, stream>>>(key, x16, (int)nX);
  cvt_f32_to_f16<<<cvtg, cvtb, 0, stream>>>(Wk, w16, (int)nW);
  proj_gemm16<<<pgrid, pblk, 0, stream>>>(x16, w16, bk, (void*)k16, 0, 1.0f);

  cvt_f32_to_f16<<<cvtg, cvtb, 0, stream>>>(value, x16, (int)nX);
  cvt_f32_to_f16<<<cvtg, cvtb, 0, stream>>>(Wv, w16, (int)nW);
  proj_gemm16<<<pgrid, pblk, 0, stream>>>(x16, w16, bv, (void*)vT16, 1, 1.0f);

  flash_attn<<<agrid, ablk, 0, stream>>>(q16, k16, vT16, mask, a16);

  cvt_f32_to_f16<<<cvtg, cvtb, 0, stream>>>(Wo, w16, (int)nW);
  proj_gemm16<<<pgrid, pblk, 0, stream>>>(a16, w16, bo, d_out, 2, 1.0f);
}
